// MessagePassingFlow_45208825757707
// MI455X (gfx1250) — matrix operand order repaired and checked by reading, not compiled, not run on hardware
//
#include <hip/hip_runtime.h>
#include <hip/hip_bf16.h>

#define NN 50000
#define NE 600000
#define DD 128
#define WW 16

typedef __attribute__((ext_vector_type(16))) __bf16 bf16x16;
typedef __attribute__((ext_vector_type(8)))  __bf16 bf16x8;
typedef __attribute__((ext_vector_type(8)))  float  f32x8;

__device__ __forceinline__ f32x8 wmma_bf16(bf16x16 a, bf16x16 b, f32x8 c) {
  return __builtin_amdgcn_wmma_f32_16x16x32_bf16(false, a, false, b, (short)0, c, false, false);
}

__device__ __forceinline__ float sigmoidf(float z) {
  return 1.0f / (1.0f + __expf(-z));
}

// ---------------------------------------------------------------------------
// Pack all weight matrices into bf16, transposed to [n][k] so a WMMA B-fragment
// (column n, 16 consecutive k) is one contiguous 32B load per lane.
// Layout in pw (bf16 elements):
//   [    0,16384) msg_w1T   [n*128+k]
//   [16384,32768) msg_w2T
//   [32768,49152) node_w1T
//   [49152,65536) node_w2T
//   [65536,73728) inv_w1T   [n*64+k], k<48 real, rest zero-padded
//   [73728,75776) gate_wT   [n*128+k], n<16
// ---------------------------------------------------------------------------
__global__ void pack_weights_kernel(const float* __restrict__ msg_w1,
                                    const float* __restrict__ msg_w2,
                                    const float* __restrict__ node_w1,
                                    const float* __restrict__ node_w2,
                                    const float* __restrict__ inv_w1,
                                    const float* __restrict__ gate_w,
                                    __bf16* __restrict__ pw) {
  int stride = gridDim.x * blockDim.x;
  for (int idx = blockIdx.x * blockDim.x + threadIdx.x; idx < 75776; idx += stride) {
    float v;
    if (idx < 65536) {
      int which = idx >> 14;
      int r = idx & 16383;
      int n = r >> 7, k = r & 127;
      const float* w = (which == 0) ? msg_w1 : (which == 1) ? msg_w2
                     : (which == 2) ? node_w1 : node_w2;
      v = w[k * DD + n];
    } else if (idx < 73728) {
      int r = idx - 65536;
      int n = r >> 6, k = r & 63;
      v = (k < 48) ? inv_w1[k * DD + n] : 0.0f;
    } else {
      int r = idx - 73728;
      int n = r >> 7, k = r & 127;
      v = gate_w[k * WW + n];
    }
    pw[idx] = (__bf16)v;
  }
}

__global__ void zero_kernel(float4* __restrict__ p, int n4) {
  int i = blockIdx.x * blockDim.x + threadIdx.x;
  if (i < n4) p[i] = make_float4(0.f, 0.f, 0.f, 0.f);
}

// ---------------------------------------------------------------------------
// alpha[e] = sigmoid(relu(ei @ inv_w1 + b1) @ inv_w2 + b2), ei = [ns,nd,cos] (E,48)
// One wave handles 16 edges; K padded 48->64 (2 WMMA K-chunks).
// 600000/16 = 37500 tiles; 4 waves/block -> 9375 blocks exactly.
// ---------------------------------------------------------------------------
__global__ void __launch_bounds__(128) alpha_kernel(
    const float* __restrict__ V, const int* __restrict__ eidx,
    const __bf16* __restrict__ iwT, const float* __restrict__ inv_b1,
    const float* __restrict__ inv_w2, const float* __restrict__ inv_b2,
    float* __restrict__ alpha) {
  __shared__ __align__(16) float ei[4][16][68];   // padded stride vs 64 banks
  int lane = threadIdx.x & 31, wv = threadIdx.x >> 5;
  int e0 = (blockIdx.x * 4 + wv) * 16;

  // Phase 1: edge invariants. lane = (edge m, half h): h covers w=8h..8h+7.
  {
    int m = lane & 15, h = lane >> 4;
    int e = e0 + m;
    int r = eidx[e], c = eidx[NE + e];
    const float* Vr = V + (size_t)r * 48;
    const float* Vc = V + (size_t)c * 48;
#pragma unroll
    for (int j = 0; j < 8; ++j) {
      int w = h * 8 + j;
      float a0 = Vr[w], a1 = Vr[16 + w], a2 = Vr[32 + w];
      float b0 = Vc[w], b1v = Vc[16 + w], b2v = Vc[32 + w];
      float ns = sqrtf(a0 * a0 + a1 * a1 + a2 * a2);
      float nd = sqrtf(b0 * b0 + b1v * b1v + b2v * b2v);
      float dt = a0 * b0 + a1 * b1v + a2 * b2v;
      ei[wv][m][w]      = ns;
      ei[wv][m][16 + w] = nd;
      ei[wv][m][32 + w] = dt / (ns * nd + 1e-8f);
      ei[wv][m][48 + w] = 0.0f;
    }
  }
  __syncthreads();

  int row = lane & 15;
  int kb = (lane < 16) ? 0 : 16;   // A-fragment K sub-offset
  int nb = (lane < 16) ? 0 : 16;   // B-fragment K sub-offset
  bf16x16 afr[2];
#pragma unroll
  for (int c = 0; c < 2; ++c) {
#pragma unroll
    for (int j = 0; j < 8; ++j) {
      afr[c][j]     = (__bf16)ei[wv][row][32 * c + kb + j];
      afr[c][8 + j] = (__bf16)ei[wv][row][32 * c + kb + 8 + j];
    }
  }

  float part[8];
#pragma unroll
  for (int r2 = 0; r2 < 8; ++r2) part[r2] = 0.0f;
#pragma unroll
  for (int t = 0; t < 8; ++t) {
    f32x8 acc = {};
#pragma unroll
    for (int c = 0; c < 2; ++c) {
      bf16x16 bfr = *(const bf16x16*)(iwT + (t * 16 + row) * 64 + 32 * c + nb);
      acc = wmma_bf16(afr[c], bfr, acc);
    }
    int n = t * 16 + row;
    float w2v = inv_w2[n];
    float b1v = inv_b1[n];
#pragma unroll
    for (int r2 = 0; r2 < 8; ++r2) {
      float h = acc[r2] + b1v;
      part[r2] += (h > 0.0f ? h : 0.0f) * w2v;
    }
  }
  // reduce over the 16 lanes that share each row-set
#pragma unroll
  for (int off = 1; off < 16; off <<= 1) {
#pragma unroll
    for (int r2 = 0; r2 < 8; ++r2) part[r2] += __shfl_xor(part[r2], off, 32);
  }
  if ((lane & 15) == 0) {
    int mbase = (lane < 16) ? 0 : 8;
    float bb = inv_b2[0];
#pragma unroll
    for (int r2 = 0; r2 < 8; ++r2)
      alpha[e0 + mbase + r2] = sigmoidf(part[r2] + bb);
  }
}

// ---------------------------------------------------------------------------
// Edge kernel: LayerNorm + msg MLP (bf16 WMMA) + residual, scatter-add into m_i.
// One wave per 16-edge tile; 4 waves/block -> 9375 blocks exactly.
// ---------------------------------------------------------------------------
__global__ void __launch_bounds__(128) edge_kernel(
    const float* __restrict__ M, const int* __restrict__ eidx,
    const float* __restrict__ ln_g, const float* __restrict__ ln_b,
    const __bf16* __restrict__ w1T, const float* __restrict__ b1,
    const __bf16* __restrict__ w2T, const float* __restrict__ b2,
    const float* __restrict__ alpha,
    float* __restrict__ M2out, float* mi) {
  __shared__ __align__(16) __bf16 h1s[4][16][136];  // stride 136: aligned + no bank conflicts
  __shared__ float s_mu[4][16];
  __shared__ float s_rs[4][16];
  int lane = threadIdx.x & 31, wv = threadIdx.x >> 5;
  int e0 = (blockIdx.x * 4 + wv) * 16;
  int row = lane & 15;
  int kb = (lane < 16) ? 0 : 16;
  int nb = (lane < 16) ? 0 : 16;
  int mb = (lane < 16) ? 0 : 8;

  // Phase A: each lane loads exactly the 64 columns of its row that the
  // 16x32 bf16 A-fragment layout needs (lanes L and L+16 split row L).
  float mv[4][16];
  const float* Mr = M + (size_t)(e0 + row) * DD;
  float sum = 0.f, ss = 0.f;
#pragma unroll
  for (int c = 0; c < 4; ++c) {
#pragma unroll
    for (int j = 0; j < 8; ++j) {
      float v0 = Mr[32 * c + kb + j];
      float v1 = Mr[32 * c + kb + 8 + j];
      mv[c][j] = v0; mv[c][8 + j] = v1;
      sum += v0 + v1;
      ss += v0 * v0 + v1 * v1;
    }
  }
  sum += __shfl_xor(sum, 16, 32);
  ss  += __shfl_xor(ss, 16, 32);
  float mu = sum * (1.0f / 128.0f);
  float var = ss * (1.0f / 128.0f) - mu * mu;
  float rs = rsqrtf(var + 1e-5f);
  if (lane < 16) { s_mu[wv][row] = mu; s_rs[wv][row] = rs; }

  bf16x16 afr[4];
#pragma unroll
  for (int c = 0; c < 4; ++c) {
#pragma unroll
    for (int j = 0; j < 8; ++j) {
      int c0 = 32 * c + kb + j, c1 = c0 + 8;
      afr[c][j]     = (__bf16)((mv[c][j]     - mu) * rs * ln_g[c0] + ln_b[c0]);
      afr[c][8 + j] = (__bf16)((mv[c][8 + j] - mu) * rs * ln_g[c1] + ln_b[c1]);
    }
  }

  // GEMM1: Mn @ w1 -> relu -> bf16 LDS staging (C-layout write)
#pragma unroll
  for (int t = 0; t < 8; ++t) {
    f32x8 acc = {};
#pragma unroll
    for (int c = 0; c < 4; ++c) {
      bf16x16 bfr = *(const bf16x16*)(w1T + (t * 16 + row) * DD + 32 * c + nb);
      acc = wmma_bf16(afr[c], bfr, acc);
    }
    int n = t * 16 + row;
    float b1v = b1[n];
#pragma unroll
    for (int r2 = 0; r2 < 8; ++r2) {
      float h = acc[r2] + b1v;
      h1s[wv][mb + r2][n] = (__bf16)(h > 0.0f ? h : 0.0f);
    }
  }
  __syncthreads();

  // Rebuild A-fragments from staged H1 (aligned 16B LDS vector loads)
  bf16x16 af2[4];
#pragma unroll
  for (int c = 0; c < 4; ++c) {
    bf16x8 lo = *(const bf16x8*)&h1s[wv][row][32 * c + kb];
    bf16x8 hi = *(const bf16x8*)&h1s[wv][row][32 * c + kb + 8];
#pragma unroll
    for (int j = 0; j < 8; ++j) { af2[c][j] = lo[j]; af2[c][8 + j] = hi[j]; }
  }

  // GEMM2 + residual + scatter
#pragma unroll
  for (int t = 0; t < 8; ++t) {
    f32x8 acc = {};
#pragma unroll
    for (int c = 0; c < 4; ++c) {
      bf16x16 bfr = *(const bf16x16*)(w2T + (t * 16 + row) * DD + 32 * c + nb);
      acc = wmma_bf16(af2[c], bfr, acc);
    }
    int n = t * 16 + row;
    float gn = ln_g[n], bn = ln_b[n], b2v = b2[n];
#pragma unroll
    for (int r2 = 0; r2 < 8; ++r2) {
      int m = mb + r2;
      int e = e0 + m;
      float mn = (M[(size_t)e * DD + n] - s_mu[wv][m]) * s_rs[wv][m] * gn + bn;
      float val = mn + acc[r2] + b2v;
      M2out[(size_t)e * DD + n] = val;
      float wgt = val * alpha[e];
      int dst = eidx[e];  // edge_index[0] = row
      __hip_atomic_fetch_add(&mi[(size_t)dst * DD + n], wgt,
                             __ATOMIC_RELAXED, __HIP_MEMORY_SCOPE_AGENT);
    }
  }
}

// ---------------------------------------------------------------------------
// Node kernel: x2 = x + relu(m_i@w1+b1)@w2+b2 ; gates = sigmoid(x2@gw+gb); V2=V*g
// m_i lives in the x2 output region (read-then-overwrite, tile-private rows).
// 5 waves/block, 3125 tiles / 5 = 625 blocks exactly.
// ---------------------------------------------------------------------------
__global__ void __launch_bounds__(160) node_kernel(
    const float* __restrict__ x, const float* __restrict__ V,
    const __bf16* __restrict__ nw1T, const float* __restrict__ nb1,
    const __bf16* __restrict__ nw2T, const float* __restrict__ nb2,
    const __bf16* __restrict__ gwT, const float* __restrict__ gb,
    float* xio, float* __restrict__ V2out) {
  __shared__ __align__(16) __bf16 h1s[5][16][136];
  __shared__ __align__(16) __bf16 x2s[5][16][136];
  int lane = threadIdx.x & 31, wv = threadIdx.x / 32;
  int n0 = (blockIdx.x * 5 + wv) * 16;
  int row = lane & 15;
  int kb = (lane < 16) ? 0 : 16;
  int nb = (lane < 16) ? 0 : 16;
  int mb = (lane < 16) ? 0 : 8;

  // A-fragments straight from m_i (fp32 -> bf16)
  const float* mr = xio + (size_t)(n0 + row) * DD;
  bf16x16 afr[4];
#pragma unroll
  for (int c = 0; c < 4; ++c) {
#pragma unroll
    for (int j = 0; j < 8; ++j) {
      afr[c][j]     = (__bf16)mr[32 * c + kb + j];
      afr[c][8 + j] = (__bf16)mr[32 * c + kb + 8 + j];
    }
  }

#pragma unroll
  for (int t = 0; t < 8; ++t) {
    f32x8 acc = {};
#pragma unroll
    for (int c = 0; c < 4; ++c) {
      bf16x16 bfr = *(const bf16x16*)(nw1T + (t * 16 + row) * DD + 32 * c + nb);
      acc = wmma_bf16(afr[c], bfr, acc);
    }
    int n = t * 16 + row;
    float b1v = nb1[n];
#pragma unroll
    for (int r2 = 0; r2 < 8; ++r2) {
      float h = acc[r2] + b1v;
      h1s[wv][mb + r2][n] = (__bf16)(h > 0.0f ? h : 0.0f);
    }
  }
  __syncthreads();

  bf16x16 af2[4];
#pragma unroll
  for (int c = 0; c < 4; ++c) {
    bf16x8 lo = *(const bf16x8*)&h1s[wv][row][32 * c + kb];
    bf16x8 hi = *(const bf16x8*)&h1s[wv][row][32 * c + kb + 8];
#pragma unroll
    for (int j = 0; j < 8; ++j) { af2[c][j] = lo[j]; af2[c][8 + j] = hi[j]; }
  }

#pragma unroll
  for (int t = 0; t < 8; ++t) {
    f32x8 acc = {};
#pragma unroll
    for (int c = 0; c < 4; ++c) {
      bf16x16 bfr = *(const bf16x16*)(nw2T + (t * 16 + row) * DD + 32 * c + nb);
      acc = wmma_bf16(af2[c], bfr, acc);
    }
    int n = t * 16 + row;
    float b2v = nb2[n];
#pragma unroll
    for (int r2 = 0; r2 < 8; ++r2) {
      int m = mb + r2;
      int node = n0 + m;
      float x2v = x[(size_t)node * DD + n] + acc[r2] + b2v;
      x2s[wv][m][n] = (__bf16)x2v;
      xio[(size_t)node * DD + n] = x2v;   // overwrite m_i with x2 output
    }
  }
  __syncthreads();

  // Gate GEMM: one 16-wide N-tile against packed gate_wT
  bf16x16 ag[4];
#pragma unroll
  for (int c = 0; c < 4; ++c) {
    bf16x8 lo = *(const bf16x8*)&x2s[wv][row][32 * c + kb];
    bf16x8 hi = *(const bf16x8*)&x2s[wv][row][32 * c + kb + 8];
#pragma unroll
    for (int j = 0; j < 8; ++j) { ag[c][j] = lo[j]; ag[c][8 + j] = hi[j]; }
  }
  f32x8 accg = {};
#pragma unroll
  for (int c = 0; c < 4; ++c) {
    bf16x16 bfr = *(const bf16x16*)(gwT + row * DD + 32 * c + nb);
    accg = wmma_bf16(ag[c], bfr, accg);
  }
  int w = row;
  float gbv = gb[w];
#pragma unroll
  for (int r2 = 0; r2 < 8; ++r2) {
    int node = n0 + mb + r2;
    float g = sigmoidf(accg[r2] + gbv);
#pragma unroll
    for (int a = 0; a < 3; ++a) {
      size_t off = (size_t)node * 48 + a * 16 + w;
      V2out[off] = V[off] * g;
    }
  }
}

extern "C" void kernel_launch(void* const* d_in, const int* in_sizes, int n_in,
                              void* d_out, int out_size, void* d_ws, size_t ws_size,
                              hipStream_t stream) {
  (void)in_sizes; (void)n_in; (void)out_size; (void)ws_size;
  const float* x       = (const float*)d_in[0];
  const float* M       = (const float*)d_in[1];
  const float* V       = (const float*)d_in[2];
  const int*   eidx    = (const int*)d_in[3];
  const float* ln_g    = (const float*)d_in[4];
  const float* ln_b    = (const float*)d_in[5];
  const float* msg_w1  = (const float*)d_in[6];
  const float* msg_b1  = (const float*)d_in[7];
  const float* msg_w2  = (const float*)d_in[8];
  const float* msg_b2  = (const float*)d_in[9];
  const float* node_w1 = (const float*)d_in[10];
  const float* node_b1 = (const float*)d_in[11];
  const float* node_w2 = (const float*)d_in[12];
  const float* node_b2 = (const float*)d_in[13];
  const float* inv_w1  = (const float*)d_in[14];
  const float* inv_b1  = (const float*)d_in[15];
  const float* inv_w2  = (const float*)d_in[16];
  const float* inv_b2  = (const float*)d_in[17];
  const float* gate_w  = (const float*)d_in[18];
  const float* gate_b  = (const float*)d_in[19];

  float* out = (float*)d_out;
  float* x2o = out;                              // 50000*128 (doubles as m_i)
  float* M2o = out + (size_t)NN * DD;            // 600000*128
  float* V2o = M2o + (size_t)NE * DD;            // 50000*48

  float*  alpha = (float*)d_ws;
  __bf16* pw    = (__bf16*)((char*)d_ws + (size_t)NE * sizeof(float));
  const __bf16* w1T  = pw;
  const __bf16* w2T  = pw + 16384;
  const __bf16* nw1T = pw + 32768;
  const __bf16* nw2T = pw + 49152;
  const __bf16* iwT  = pw + 65536;
  const __bf16* gwT  = pw + 73728;

  pack_weights_kernel<<<296, 256, 0, stream>>>(msg_w1, msg_w2, node_w1, node_w2,
                                               inv_w1, gate_w, pw);
  zero_kernel<<<(NN * DD / 4 + 255) / 256, 256, 0, stream>>>((float4*)x2o, NN * DD / 4);
  alpha_kernel<<<9375, 128, 0, stream>>>(V, eidx, iwT, inv_b1, inv_w2, inv_b2, alpha);
  edge_kernel<<<9375, 128, 0, stream>>>(M, eidx, ln_g, ln_b, w1T, msg_b1,
                                        w2T, msg_b2, alpha, M2o, x2o);
  node_kernel<<<625, 160, 0, stream>>>(x, V, nw1T, node_b1, nw2T, node_b2,
                                       gwT, gate_b, x2o, V2o);
}
